// P6MoeBlock_773094113558
// MI455X (gfx1250) — compile-verified, matrix operand order checked by reading, not run on hardware
//
#include <hip/hip_runtime.h>
#include <hip/hip_bf16.h>
#include <stdint.h>

// ---------------- problem constants ----------------
#define T_TOKENS 8192
#define H_DIM    2048
#define FFN_DIM  5632
#define N_EXP    8
#define TOTAL_SLOTS (T_TOKENS * 2)

// LDS row pitch for a 32 x bf16 (64B) row, padded to 80B (keeps 16B alignment,
// spreads rows across the 64 LDS banks).
#define LDS_PITCH 80
#define A_BUF_BYTES (128 * LDS_PITCH)
#define B_BUF_BYTES (64 * LDS_PITCH)

typedef __attribute__((ext_vector_type(16))) __bf16 v16bf;
typedef __attribute__((ext_vector_type(8)))  __bf16 v8bf;
typedef __attribute__((ext_vector_type(8)))  float  v8f;

union FragBF { v16bf v; v8bf h[2]; };

// Load one 16x32 bf16 WMMA A/B fragment from a row-major LDS tile.
// ISA layout (CDNA5 ISA): lane m (0..15) holds row m; lanes<16 K={0..7,16..23},
// lanes>=16 K={8..15,24..31} -> two 16B chunks at byte offsets half*16 and 32+half*16.
__device__ inline v16bf load_frag(const char* base, int row, int half) {
    FragBF f;
    const char* p = base + row * LDS_PITCH + half * 16;
    f.h[0] = *(const v8bf*)(p);
    f.h[1] = *(const v8bf*)(p + 32);
    return f.v;
}

// convert 16 f32 (in 4 float4 regs) -> 16 bf16 into LDS (two 16B chunks)
__device__ inline void cvt_regs16(const float4* f, char* dst) {
    v8bf o0, o1;
    o0[0]=(__bf16)f[0].x; o0[1]=(__bf16)f[0].y; o0[2]=(__bf16)f[0].z; o0[3]=(__bf16)f[0].w;
    o0[4]=(__bf16)f[1].x; o0[5]=(__bf16)f[1].y; o0[6]=(__bf16)f[1].z; o0[7]=(__bf16)f[1].w;
    o1[0]=(__bf16)f[2].x; o1[1]=(__bf16)f[2].y; o1[2]=(__bf16)f[2].z; o1[3]=(__bf16)f[2].w;
    o1[4]=(__bf16)f[3].x; o1[5]=(__bf16)f[3].y; o1[6]=(__bf16)f[3].z; o1[7]=(__bf16)f[3].w;
    *(v8bf*)dst        = o0;
    *(v8bf*)(dst + 16) = o1;
}

// convert 8 f32 (in 2 float4 regs) -> 8 bf16 into LDS (one 16B chunk)
__device__ inline void cvt_regs8(const float4* f, char* dst) {
    v8bf o;
    o[0]=(__bf16)f[0].x; o[1]=(__bf16)f[0].y; o[2]=(__bf16)f[0].z; o[3]=(__bf16)f[0].w;
    o[4]=(__bf16)f[1].x; o[5]=(__bf16)f[1].y; o[6]=(__bf16)f[1].z; o[7]=(__bf16)f[1].w;
    *(v8bf*)dst = o;
}

// CDNA5 async global->LDS raw copy of 32B (two B128 ops), tracked by ASYNCcnt.
// Per ISA 15.18.3: the instruction offset applies to both the LDS and the global address.
__device__ inline void async_copy32(uint32_t lds_addr, const void* gaddr) {
    asm volatile(
        "global_load_async_to_lds_b128 %0, %1, off\n\t"
        "global_load_async_to_lds_b128 %0, %1, off offset:16"
        :: "v"(lds_addr), "v"(gaddr) : "memory");
}
__device__ inline void wait_asynccnt0() {
    asm volatile("s_wait_asynccnt 0x0" ::: "memory");
}

// silu(a)*b with hardware v_rcp_f32 (result is rounded to bf16 anyway)
__device__ inline float silu_mul(float a, float b) {
    return a * __builtin_amdgcn_rcpf(1.f + __expf(-a)) * b;
}

// ---------------- small kernels ----------------
__global__ void init_kernel(int* counts, int* cursor) {
    int i = threadIdx.x;
    if (i < N_EXP) { counts[i] = 0; cursor[i] = 0; }
}

__global__ void wrun_kernel(const float* __restrict__ wg,
                            const float* __restrict__ wg_ema,
                            float* __restrict__ wrun) {
    int i = blockIdx.x * 256 + threadIdx.x;
    if (i < H_DIM * N_EXP) wrun[i] = 0.5f * (wg[i] + wg_ema[i]);
}

// one wave32 per token
__global__ __launch_bounds__(256)
void router_kernel(const float* __restrict__ x, const float* __restrict__ wrun,
                   float* __restrict__ logits, int* __restrict__ counts,
                   int* __restrict__ topk_e, float* __restrict__ topk_w) {
    int t = blockIdx.x * 8 + (threadIdx.x >> 5);
    int lane = threadIdx.x & 31;
    const float* xr = x + (size_t)t * H_DIM;
    float acc[N_EXP];
#pragma unroll
    for (int e = 0; e < N_EXP; e++) acc[e] = 0.f;
    for (int h = lane; h < H_DIM; h += 32) {
        float xv = xr[h];
        const float* wr = wrun + h * N_EXP;
#pragma unroll
        for (int e = 0; e < N_EXP; e++) acc[e] += xv * wr[e];
    }
#pragma unroll
    for (int off = 16; off >= 1; off >>= 1)
#pragma unroll
        for (int e = 0; e < N_EXP; e++) acc[e] += __shfl_xor(acc[e], off, 32);
    if (lane == 0) {
        float mx = acc[0];
#pragma unroll
        for (int e = 1; e < N_EXP; e++) mx = fmaxf(mx, acc[e]);
        float p[N_EXP];
#pragma unroll
        for (int e = 0; e < N_EXP; e++) p[e] = __expf(acc[e] - mx);
        int e0 = 0;
#pragma unroll
        for (int e = 1; e < N_EXP; e++) if (p[e] > p[e0]) e0 = e;
        int e1 = (e0 == 0) ? 1 : 0;
#pragma unroll
        for (int e = 0; e < N_EXP; e++) if (e != e0 && p[e] > p[e1]) e1 = e;
        float denom = p[e0] + p[e1];
        topk_e[2 * t]     = e0;
        topk_e[2 * t + 1] = e1;
        topk_w[2 * t]     = p[e0] / denom;
        topk_w[2 * t + 1] = p[e1] / denom;
#pragma unroll
        for (int e = 0; e < N_EXP; e++) logits[(size_t)t * N_EXP + e] = acc[e];
        atomicAdd(&counts[e0], 1);
        atomicAdd(&counts[e1], 1);
    }
}

__global__ void scan_kernel(const int* counts, int* basep, int* cursor) {
    if (threadIdx.x == 0) {
        int s = 0;
        for (int e = 0; e < N_EXP; e++) { basep[e] = s; s += counts[e]; }
    }
    if (threadIdx.x < N_EXP) cursor[threadIdx.x] = 0;
}

__global__ void scatter_kernel(const int* __restrict__ topk_e,
                               const int* __restrict__ basep, int* cursor,
                               int* __restrict__ expert_tok,
                               int* __restrict__ slotA, int* __restrict__ slotB) {
    int t = blockIdx.x * 256 + threadIdx.x;
    if (t >= T_TOKENS) return;
    int e0 = topk_e[2 * t], e1 = topk_e[2 * t + 1];
    int s0 = basep[e0] + atomicAdd(&cursor[e0], 1);
    expert_tok[s0] = t;
    slotA[t] = s0;
    int s1 = basep[e1] + atomicAdd(&cursor[e1], 1);
    expert_tok[s1] = t;
    slotB[t] = s1;
}

// ---------------- GEMM 1: gathered x @ fc1_1^T / fc1_2^T, silu*mul -> G (bf16) ----------------
// block tile: 128 tokens x 64 ffn, 8 waves as 4(M) x 2(N), each wave 32x32 per matrix.
// Double-buffered LDS: global loads for stage k+1 issue before the WMMAs of stage k.
__global__ __launch_bounds__(256)
void fc1_glu_kernel(const float* __restrict__ x,
                    const float* __restrict__ fc11,
                    const float* __restrict__ fc12,
                    const int* __restrict__ counts,
                    const int* __restrict__ basep,
                    const int* __restrict__ expert_tok,
                    __bf16* __restrict__ Gbuf) {
    const int e  = blockIdx.z;
    const int n  = counts[e];
    const int mt = blockIdx.y;
    if (mt * 128 >= n) return;
    const int nt   = blockIdx.x;
    const int base = basep[e];
    const int tid  = threadIdx.x;

    __shared__ __align__(16) char sA[2 * A_BUF_BYTES];
    __shared__ __align__(16) char sB1[2 * B_BUF_BYTES];
    __shared__ __align__(16) char sB2[2 * B_BUF_BYTES];
    __shared__ int sTok[128];

    if (tid < 128) {
        int rloc = mt * 128 + tid;
        sTok[tid] = expert_tok[base + (rloc < n ? rloc : 0)];
    }
    __syncthreads();

    const int w = tid >> 5, lane = tid & 31;
    const int wm = w & 3, wn = w >> 2;
    const int half = lane >> 4, l15 = lane & 15;

    // A loader: 128 rows x 2 half-rows (16 f32 each)
    const int ar = tid >> 1, ah = tid & 1;
    const int atok = sTok[ar];
    const float* asrc = x + (size_t)atok * H_DIM + ah * 16;
    char* adst0 = sA + ar * LDS_PITCH + ah * 32;

    // B loader: 2 matrices x 64 rows x 2 half-rows (16 f32 each)
    const int bm = tid >> 7;
    const int br = (tid >> 1) & 63;
    const size_t wofs = ((size_t)e * FFN_DIM + (size_t)nt * 64 + br) * H_DIM + ah * 16;
    const float* bsrc = (bm ? fc12 : fc11) + wofs;
    char* bdst0 = (bm ? sB2 : sB1) + br * LDS_PITCH + ah * 32;

    v8f acc1[2][2] = {};
    v8f acc2[2][2] = {};

    float4 aR[4], bR[4];
    // prime stage 0
    {
        const float4* a4 = (const float4*)asrc;
        const float4* b4 = (const float4*)bsrc;
#pragma unroll
        for (int i = 0; i < 4; i++) { aR[i] = a4[i]; bR[i] = b4[i]; }
        cvt_regs16(aR, adst0);
        cvt_regs16(bR, bdst0);
    }
    __syncthreads();

    int buf = 0;
    for (int k0 = 0; k0 < H_DIM; k0 += 32) {
        const bool has_next = (k0 + 32 < H_DIM);
        if (has_next) {
            const float4* a4 = (const float4*)(asrc + k0 + 32);
            const float4* b4 = (const float4*)(bsrc + k0 + 32);
#pragma unroll
            for (int i = 0; i < 4; i++) { aR[i] = a4[i]; bR[i] = b4[i]; }
        }

        const char* pA  = sA  + buf * A_BUF_BYTES;
        const char* pB1 = sB1 + buf * B_BUF_BYTES;
        const char* pB2 = sB2 + buf * B_BUF_BYTES;

        v16bf afrag[2], b1frag[2], b2frag[2];
#pragma unroll
        for (int i = 0; i < 2; i++) {
            afrag[i]  = load_frag(pA,  wm * 32 + i * 16 + l15, half);
            b1frag[i] = load_frag(pB1, wn * 32 + i * 16 + l15, half);
            b2frag[i] = load_frag(pB2, wn * 32 + i * 16 + l15, half);
        }
#pragma unroll
        for (int mi = 0; mi < 2; mi++)
#pragma unroll
            for (int ni = 0; ni < 2; ni++) {
                acc1[mi][ni] = __builtin_amdgcn_wmma_f32_16x16x32_bf16(
                    false, afrag[mi], false, b1frag[ni], (short)0, acc1[mi][ni], false, false);
                acc2[mi][ni] = __builtin_amdgcn_wmma_f32_16x16x32_bf16(
                    false, afrag[mi], false, b2frag[ni], (short)0, acc2[mi][ni], false, false);
            }

        if (has_next) {
            const int nb = buf ^ 1;
            cvt_regs16(aR, adst0 + nb * A_BUF_BYTES);
            cvt_regs16(bR, bdst0 + nb * B_BUF_BYTES);
        }
        __syncthreads();
        buf ^= 1;
    }

    // epilogue: g = silu(h1) * h2 -> bf16 G
    const bool full = (mt * 128 + 128 <= n);
#pragma unroll
    for (int mi = 0; mi < 2; mi++)
#pragma unroll
        for (int ni = 0; ni < 2; ni++) {
            v8f h1 = acc1[mi][ni], h2 = acc2[mi][ni];
            const int colg  = nt * 64 + wn * 32 + ni * 16 + l15;
            const int rloc0 = mt * 128 + wm * 32 + mi * 16 + 8 * half;
            __bf16* gp = Gbuf + (size_t)(base + rloc0) * FFN_DIM + colg;
            if (full) {
#pragma unroll
                for (int r = 0; r < 8; r++) {
                    *gp = (__bf16)silu_mul(h1[r], h2[r]);
                    gp += FFN_DIM;
                }
            } else {
#pragma unroll
                for (int r = 0; r < 8; r++) {
                    if (rloc0 + r < n) *gp = (__bf16)silu_mul(h1[r], h2[r]);
                    gp += FFN_DIM;
                }
            }
        }
}

// ---------------- GEMM 2: G (bf16) @ fc2^T -> Y (f32) ----------------
// A tile (raw bf16) is staged with CDNA5 async global->LDS copies (ASYNCcnt);
// B tile (f32 weights) goes through VGPRs for the bf16 conversion.
__global__ __launch_bounds__(256)
void fc2_kernel(const __bf16* __restrict__ Gbuf,
                const float* __restrict__ fc2w,
                const int* __restrict__ counts,
                const int* __restrict__ basep,
                float* __restrict__ Ybuf) {
    const int e  = blockIdx.z;
    const int n  = counts[e];
    const int mt = blockIdx.y;
    if (mt * 128 >= n) return;
    const int nt   = blockIdx.x;   // H tile of 64
    const int base = basep[e];
    const int tid  = threadIdx.x;

    __shared__ __align__(16) char sA[2 * A_BUF_BYTES];
    __shared__ __align__(16) char sB[2 * B_BUF_BYTES];

    const int w = tid >> 5, lane = tid & 31;
    const int wm = w & 3, wn = w >> 2;
    const int half = lane >> 4, l15 = lane & 15;

    // A loader: bf16 rows from Gbuf (already gathered/grouped), async raw 32B copies
    const int ar = tid >> 1, ah = tid & 1;
    int arow = mt * 128 + ar;
    if (arow >= n) arow = 0;  // clamp to a valid row; result masked at store
    const __bf16* asrc = Gbuf + (size_t)(base + arow) * FFN_DIM + ah * 16;
    char* adst0 = sA + ar * LDS_PITCH + ah * 32;
    const uint32_t alds0 = (uint32_t)(uintptr_t)adst0;  // LDS addr = generic addr[31:0]

    // B loader: 64 rows x 32 cols f32 -> bf16 ; 256 threads x 8 floats
    const int br = tid >> 2, bq = tid & 3;
    const float* bsrc = fc2w + ((size_t)e * H_DIM + (size_t)nt * 64 + br) * FFN_DIM + bq * 8;
    char* bdst0 = sB + br * LDS_PITCH + bq * 16;

    v8f acc[2][2] = {};

    float4 bR[2];
    {
        async_copy32(alds0, asrc);                  // A stage 0, global -> LDS direct
        const float4* b4 = (const float4*)bsrc;
        bR[0] = b4[0]; bR[1] = b4[1];
        cvt_regs8(bR, bdst0);
        wait_asynccnt0();
    }
    __syncthreads();

    int buf = 0;
    for (int k0 = 0; k0 < FFN_DIM; k0 += 32) {
        const bool has_next = (k0 + 32 < FFN_DIM);
        const int nb = buf ^ 1;
        if (has_next) {
            // next A tile: async copy straight into the other LDS buffer
            async_copy32(alds0 + nb * A_BUF_BYTES, asrc + k0 + 32);
            const float4* b4 = (const float4*)(bsrc + k0 + 32);
            bR[0] = b4[0]; bR[1] = b4[1];
        }

        const char* pA = sA + buf * A_BUF_BYTES;
        const char* pB = sB + buf * B_BUF_BYTES;

        v16bf afrag[2], bfrag[2];
#pragma unroll
        for (int i = 0; i < 2; i++) {
            afrag[i] = load_frag(pA, wm * 32 + i * 16 + l15, half);
            bfrag[i] = load_frag(pB, wn * 32 + i * 16 + l15, half);
        }
#pragma unroll
        for (int mi = 0; mi < 2; mi++)
#pragma unroll
            for (int ni = 0; ni < 2; ni++)
                acc[mi][ni] = __builtin_amdgcn_wmma_f32_16x16x32_bf16(
                    false, afrag[mi], false, bfrag[ni], (short)0, acc[mi][ni], false, false);

        if (has_next)
            cvt_regs8(bR, bdst0 + nb * B_BUF_BYTES);

        wait_asynccnt0();   // next-stage A tile landed in LDS
        __syncthreads();
        buf ^= 1;
    }

    const bool full = (mt * 128 + 128 <= n);
#pragma unroll
    for (int mi = 0; mi < 2; mi++)
#pragma unroll
        for (int ni = 0; ni < 2; ni++) {
            v8f d = acc[mi][ni];
            const int colg  = nt * 64 + wn * 32 + ni * 16 + l15;
            const int rloc0 = mt * 128 + wm * 32 + mi * 16 + 8 * half;
            float* yp = Ybuf + (size_t)(base + rloc0) * H_DIM + colg;
            if (full) {
#pragma unroll
                for (int r = 0; r < 8; r++) { *yp = d[r]; yp += H_DIM; }
            } else {
#pragma unroll
                for (int r = 0; r < 8; r++) {
                    if (rloc0 + r < n) *yp = d[r];
                    yp += H_DIM;
                }
            }
        }
}

// ---------------- combine: out[t] = w0*Y[slot0] + w1*Y[slot1] ----------------
__global__ __launch_bounds__(256)
void combine_kernel(const float* __restrict__ Ybuf,
                    const int* __restrict__ slotA, const int* __restrict__ slotB,
                    const float* __restrict__ topk_w, float* __restrict__ out) {
    int t = blockIdx.x;
    float w0 = topk_w[2 * t], w1 = topk_w[2 * t + 1];
    const float4* y0 = (const float4*)(Ybuf + (size_t)slotA[t] * H_DIM);
    const float4* y1 = (const float4*)(Ybuf + (size_t)slotB[t] * H_DIM);
    float4* o = (float4*)(out + (size_t)t * H_DIM);
    for (int i = threadIdx.x; i < H_DIM / 4; i += 256) {
        float4 a = y0[i], b = y1[i], r;
        r.x = w0 * a.x + w1 * b.x;
        r.y = w0 * a.y + w1 * b.y;
        r.z = w0 * a.z + w1 * b.z;
        r.w = w0 * a.w + w1 * b.w;
        o[i] = r;
    }
}

// ---------------- host launcher ----------------
extern "C" void kernel_launch(void* const* d_in, const int* in_sizes, int n_in,
                              void* d_out, int out_size, void* d_ws, size_t ws_size,
                              hipStream_t stream) {
    const float* x      = (const float*)d_in[0];   // [T,H]
    const float* wg     = (const float*)d_in[1];   // [H,E]
    const float* wg_ema = (const float*)d_in[2];   // [H,E]
    const float* fc11   = (const float*)d_in[3];   // [E,FFN,H]
    const float* fc12   = (const float*)d_in[4];   // [E,FFN,H]
    const float* fc2w   = (const float*)d_in[5];   // [E,H,FFN]

    float* out    = (float*)d_out;                       // [T,H]
    float* logits = out + (size_t)T_TOKENS * H_DIM;      // [T,E]

    // workspace layout (256B aligned chunks); total ~320 MB
    char* ws = (char*)d_ws;
    size_t o = 0;
    int*   counts     = (int*)(ws + o);   o += 256;
    int*   basep      = (int*)(ws + o);   o += 256;
    int*   cursor     = (int*)(ws + o);   o += 256;
    float* wrun       = (float*)(ws + o); o += (size_t)H_DIM * N_EXP * 4;      // 64 KB
    int*   topk_e     = (int*)(ws + o);   o += (size_t)T_TOKENS * 2 * 4;       // 64 KB
    float* topk_w     = (float*)(ws + o); o += (size_t)T_TOKENS * 2 * 4;       // 64 KB
    int*   slotA      = (int*)(ws + o);   o += (size_t)T_TOKENS * 4;           // 32 KB
    int*   slotB      = (int*)(ws + o);   o += (size_t)T_TOKENS * 4;           // 32 KB
    int*   expert_tok = (int*)(ws + o);   o += (size_t)TOTAL_SLOTS * 4;        // 64 KB
    __bf16* Gbuf      = (__bf16*)(ws + o); o += (size_t)TOTAL_SLOTS * FFN_DIM * 2; // 184 MB
    float* Ybuf       = (float*)(ws + o);  o += (size_t)TOTAL_SLOTS * H_DIM * 4;   // 134 MB
    (void)o; (void)ws_size; (void)in_sizes; (void)n_in; (void)out_size;

    init_kernel<<<1, 32, 0, stream>>>(counts, cursor);
    wrun_kernel<<<(H_DIM * N_EXP + 255) / 256, 256, 0, stream>>>(wg, wg_ema, wrun);
    router_kernel<<<T_TOKENS / 8, 256, 0, stream>>>(x, wrun, logits, counts, topk_e, topk_w);
    scan_kernel<<<1, 32, 0, stream>>>(counts, basep, cursor);
    scatter_kernel<<<T_TOKENS / 256, 256, 0, stream>>>(topk_e, basep, cursor,
                                                       expert_tok, slotA, slotB);

    // max rows per expert = T (each token hits an expert at most once) -> 64 tiles of 128
    dim3 g1(FFN_DIM / 64, T_TOKENS / 128, N_EXP);
    fc1_glu_kernel<<<g1, 256, 0, stream>>>(x, fc11, fc12, counts, basep, expert_tok, Gbuf);

    dim3 g2(H_DIM / 64, T_TOKENS / 128, N_EXP);
    fc2_kernel<<<g2, 256, 0, stream>>>(Gbuf, fc2w, counts, basep, Ybuf);

    combine_kernel<<<T_TOKENS, 256, 0, stream>>>(Ybuf, slotA, slotB, topk_w, out);
}
